// Starcoder2Attention_8486855377466
// MI455X (gfx1250) — matrix operand order repaired and checked by reading, not compiled, not run on hardware
//
#include <hip/hip_runtime.h>

typedef __attribute__((ext_vector_type(16))) __bf16 v16bf;
typedef __attribute__((ext_vector_type(8)))  __bf16 v8bf;
typedef __attribute__((ext_vector_type(4)))  __bf16 v4bf;
typedef __attribute__((ext_vector_type(8)))  float  v8f;

__device__ __forceinline__ __bf16 f2bf(float f) {
  unsigned u = __builtin_bit_cast(unsigned, f);
  unsigned r = u + 0x7FFFu + ((u >> 16) & 1u);   // round-to-nearest-even
  unsigned short h = (unsigned short)(r >> 16);
  return __builtin_bit_cast(__bf16, h);
}
__device__ __forceinline__ float bf2f(__bf16 b) {
  unsigned u = ((unsigned)__builtin_bit_cast(unsigned short, b)) << 16;
  return __builtin_bit_cast(float, u);
}
__device__ __forceinline__ void store_out(float* p, float v)  { *p = v; }
__device__ __forceinline__ void store_out(__bf16* p, float v) { *p = f2bf(v); }

// A-fragment loader: two contiguous 16B chunks at p and p+8 elements
// (contiguous k order used here for A and B: lane(half,m): a[0..15]=K(hf*16..+15))
__device__ __forceinline__ v16bf load_a16(const __bf16* p) {
  v8bf lo = *(const v8bf*)p;
  v8bf hi = *(const v8bf*)(p + 8);
  return __builtin_shufflevector(lo, hi, 0, 1, 2, 3, 4, 5, 6, 7,
                                 8, 9, 10, 11, 12, 13, 14, 15);
}

// ---------------------------------------------------------------------------
// fp32 -> bf16 bulk cast (one-time; keeps conversions out of GEMM hot loops)
// ---------------------------------------------------------------------------
__global__ void cast_kernel(const float* __restrict__ in, __bf16* __restrict__ out,
                            int n4) {
  int i = blockIdx.x * blockDim.x + threadIdx.x;
  if (i >= n4) return;
  const float* ip = in + (size_t)i * 4;
  v4bf o;
#pragma unroll
  for (int j = 0; j < 4; ++j) o[j] = f2bf(ip[j]);
  *(v4bf*)(out + (size_t)i * 4) = o;
}

// ---------------------------------------------------------------------------
// Y = X[M,K] @ W[N,K]^T + bias[N]     (X, W bf16; K % 64 == 0)
// One wave: 64x64 tile (4 m-subtiles x 4 n-subtiles, 16 accumulators) --
// balanced A/B reuse halves L2 traffic vs 32x64. Double-buffered fragments so
// loads for k+32 overlap WMMAs for k; L2 prefetch ~8 k-steps ahead.
// TR=true stores Y transposed as Y[col*M + row] (used for V so attention's
// P@V B-fragments are contiguous loads).
// ---------------------------------------------------------------------------
template <typename OUT_T, bool TR>
__global__ __launch_bounds__(128) void gemm_wmma(
    const __bf16* __restrict__ X, const __bf16* __restrict__ W,
    const float* __restrict__ bias, OUT_T* __restrict__ Y,
    int M, int N, int K) {
  const int lane = threadIdx.x & 31;
  const int wid  = threadIdx.x >> 5;
  const int hf   = lane >> 4;
  const int ln   = lane & 15;
  const int n64  = N >> 6;
  const int w    = blockIdx.x * 4 + wid;
  if (w >= (M >> 6) * n64) return;
  const int mt = w / n64;
  const int nt = w % n64;

  const __bf16* xb = X + (size_t)(mt * 64 + ln) * (size_t)K + hf * 16;
  const __bf16* wb = W + (size_t)(nt * 64 + ln) * (size_t)K + hf * 16;

  v8f acc[4][4] = {};
  v16bf a[4], b[4], a2[4], b2[4];

  auto load_set = [&](v16bf(&A)[4], v16bf(&B)[4], int kk) {
#pragma unroll
    for (int mi = 0; mi < 4; ++mi)
      A[mi] = load_a16(xb + (size_t)(mi * 16) * K + kk);
#pragma unroll
    for (int t = 0; t < 4; ++t)
      B[t] = *(const v16bf*)(wb + (size_t)(t * 16) * K + kk);
  };
  auto do_wmma = [&](v16bf(&A)[4], v16bf(&B)[4]) {
#pragma unroll
    for (int t = 0; t < 4; ++t)
#pragma unroll
      for (int mi = 0; mi < 4; ++mi)
        acc[mi][t] = __builtin_amdgcn_wmma_f32_16x16x32_bf16(
            false, A[mi], false, B[t], (short)0, acc[mi][t], false, false);
  };

  load_set(a, b, 0);
  for (int kk = 0; kk < K; kk += 64) {
    load_set(a2, b2, kk + 32);
    if (kk + 256 < K) {
      __builtin_prefetch(xb + kk + 256, 0, 1);   // global_prefetch_b8
      __builtin_prefetch(wb + kk + 256, 0, 1);
    }
    do_wmma(a, b);
    if (kk + 64 < K) load_set(a, b, kk + 64);
    do_wmma(a2, b2);
  }

#pragma unroll
  for (int t = 0; t < 4; ++t) {
    const int col = nt * 64 + t * 16 + ln;
    const float bv = bias[col];
#pragma unroll
    for (int mi = 0; mi < 4; ++mi)
#pragma unroll
      for (int r = 0; r < 8; ++r) {
        const int row = mt * 64 + mi * 16 + hf * 8 + r;
        if (TR) store_out(Y + (size_t)col * M + row, acc[mi][t][r] + bv);
        else    store_out(Y + (size_t)row * N + col, acc[mi][t][r] + bv);
      }
  }
}

// ---------------------------------------------------------------------------
// RoPE: out[s,h,d] = in[s,h,d]*cos[s,d] + rot_half(in)[s,h,d]*sin[s,d]
// ---------------------------------------------------------------------------
__global__ void rope_kernel(const __bf16* __restrict__ in, __bf16* __restrict__ out,
                            const float* __restrict__ cosT, const float* __restrict__ sinT,
                            int nheads, int total) {
  int idx = blockIdx.x * blockDim.x + threadIdx.x;
  if (idx >= total) return;
  int d = idx & 127;
  int s = (idx >> 7) / nheads;
  float c  = cosT[s * 128 + d];
  float sn = sinT[s * 128 + d];
  float x  = bf2f(in[idx]);
  float pr = (d < 64) ? -bf2f(in[idx + 64]) : bf2f(in[idx - 64]);
  out[idx] = f2bf(x * c + pr * sn);
}

// ---------------------------------------------------------------------------
// One 32-wide kv chunk of flash attention for a 16-row q tile.
// MASKED=false for interior chunks (no causal/window edge) -> no compares.
// K rows are [kv][d] (contiguous in d); V is pre-transposed [d][kv]
// (contiguous in kv), so every fragment is a 32B vector load.
// ---------------------------------------------------------------------------
template <bool MASKED>
__device__ __forceinline__ void attn_chunk(
    const __bf16* __restrict__ qrow, const __bf16* __restrict__ kh,
    const __bf16* __restrict__ vth, __bf16* sP,
    v8f (&oacc)[8], float (&mrow)[8], float (&lrow)[8],
    int q0, int kb, int window, int hf, int ln) {
  const float scale = 0.08838834764831845f;  // 128^-0.5
  // ---- S = Q Kt (two 16x16 logit tiles over the 32 kv columns) ----
  v8f s0 = {}, s1 = {};
#pragma unroll
  for (int dc = 0; dc < 4; ++dc) {
    const int d0 = dc * 32;
    v16bf a  = load_a16(qrow + d0 + hf * 16);
    v16bf b0 = *(const v16bf*)(kh + (size_t)(kb + ln) * 512 + d0 + hf * 16);
    v16bf b1 = *(const v16bf*)(kh + (size_t)(kb + 16 + ln) * 512 + d0 + hf * 16);
    s0 = __builtin_amdgcn_wmma_f32_16x16x32_bf16(false, a, false, b0, (short)0, s0, false, false);
    s1 = __builtin_amdgcn_wmma_f32_16x16x32_bf16(false, a, false, b1, (short)0, s1, false, false);
  }
  // ---- online softmax (row r lives in 16-lane group `hf`) ----
  float alpha[8];
#pragma unroll
  for (int r = 0; r < 8; ++r) {
    float sv0 = s0[r] * scale;
    float sv1 = s1[r] * scale;
    bool m0 = true, m1 = true;
    if (MASKED) {
      const int qi  = q0 + hf * 8 + r;
      const int k0i = kb + ln;
      const int k1i = k0i + 16;
      m0 = (k0i <= qi) && (qi - k0i < window);
      m1 = (k1i <= qi) && (qi - k1i < window);
      sv0 = m0 ? sv0 : -3.0e38f;
      sv1 = m1 ? sv1 : -3.0e38f;
    }
    float rm = fmaxf(sv0, sv1);
#pragma unroll
    for (int off = 1; off < 16; off <<= 1) rm = fmaxf(rm, __shfl_xor(rm, off, 16));
    const float mn = fmaxf(mrow[r], rm);
    alpha[r] = __expf(mrow[r] - mn);
    float p0 = __expf(sv0 - mn);
    float p1 = __expf(sv1 - mn);
    if (MASKED) { p0 = m0 ? p0 : 0.f; p1 = m1 ? p1 : 0.f; }
    sP[(hf * 8 + r) * 32 + ln]      = f2bf(p0);
    sP[(hf * 8 + r) * 32 + 16 + ln] = f2bf(p1);
    float ps = p0 + p1;
#pragma unroll
    for (int off = 1; off < 16; off <<= 1) ps += __shfl_xor(ps, off, 16);
    lrow[r] = lrow[r] * alpha[r] + ps;
    mrow[r] = mn;
  }
  // ---- P in A-layout via LDS (same-wave DS ops are in-order) ----
  v16bf ap = load_a16(sP + ln * 32 + hf * 16);
  // ---- O = O*alpha + P V (8 d-chunks of 16) ----
#pragma unroll
  for (int c = 0; c < 8; ++c) {
    v8f oc = oacc[c];
#pragma unroll
    for (int r = 0; r < 8; ++r) oc[r] *= alpha[r];
    v16bf bv = *(const v16bf*)(vth + (size_t)(c * 16 + ln) * 2048 + kb + hf * 16);
    oacc[c] = __builtin_amdgcn_wmma_f32_16x16x32_bf16(false, ap, false, bv, (short)0, oc, false, false);
  }
}

// ---------------------------------------------------------------------------
// Flash attention: one wave per (head, 16-row q tile), kv chunks of 32.
// ---------------------------------------------------------------------------
__global__ __launch_bounds__(32) void attn_kernel(
    const __bf16* __restrict__ Q, const __bf16* __restrict__ Kb,
    const __bf16* __restrict__ Vt, __bf16* __restrict__ O,
    const int* __restrict__ winp) {
  const int window = *winp;
  const int h   = blockIdx.y;      // 0..35
  const int qt  = blockIdx.x;      // 0..127
  const int kvh = h / 9;           // GQA: 9 Q heads per KV head
  const int lane = threadIdx.x & 31;
  const int hf = lane >> 4;
  const int ln = lane & 15;
  const int q0 = qt * 16;

  __shared__ __bf16 sP[16 * 32];

  const __bf16* qrow = Q + (size_t)(q0 + ln) * 4608 + h * 128;
  const __bf16* kh   = Kb + (size_t)kvh * 128;
  const __bf16* vth  = Vt + (size_t)kvh * 128 * 2048;  // [d][s] within head

  v8f oacc[8] = {};
  float mrow[8], lrow[8];
#pragma unroll
  for (int r = 0; r < 8; ++r) { mrow[r] = -3.0e38f; lrow[r] = 0.f; }

  const int jc_end = (q0 + 15) >> 5;
  const int mink   = q0 - (window - 1);
  const int jc0    = mink > 0 ? (mink >> 5) : 0;
  // fully-valid chunk range: kb >= q0+16-window (window) and kb+31 <= q0 (causal)
  const int lo       = q0 + 16 - window;
  const int jlo_full = lo > 0 ? ((lo + 31) >> 5) : 0;
  const int jhi_full = q0 >= 31 ? ((q0 - 31) >> 5) : -1;

  for (int jc = jc0; jc <= jc_end; ++jc) {
    if (jc >= jlo_full && jc <= jhi_full)
      attn_chunk<false>(qrow, kh, vth, sP, oacc, mrow, lrow, q0, jc << 5, window, hf, ln);
    else
      attn_chunk<true>(qrow, kh, vth, sP, oacc, mrow, lrow, q0, jc << 5, window, hf, ln);
  }
#pragma unroll
  for (int c = 0; c < 8; ++c) {
#pragma unroll
    for (int r = 0; r < 8; ++r) {
      O[(size_t)(q0 + hf * 8 + r) * 4608 + h * 128 + c * 16 + ln] =
          f2bf(oacc[c][r] / lrow[r]);
    }
  }
}

// ---------------------------------------------------------------------------
extern "C" void kernel_launch(void* const* d_in, const int* in_sizes, int n_in,
                              void* d_out, int out_size, void* d_ws, size_t ws_size,
                              hipStream_t stream) {
  (void)in_sizes; (void)n_in; (void)out_size; (void)ws_size;
  const float* hs   = (const float*)d_in[0];
  // d_in[1] = position_ids (arange; row index used directly)
  const float* cosT = (const float*)d_in[2];
  const float* sinT = (const float*)d_in[3];
  const float* q_w  = (const float*)d_in[4];
  const float* q_b  = (const float*)d_in[5];
  const float* k_w  = (const float*)d_in[6];
  const float* k_b  = (const float*)d_in[7];
  const float* v_w  = (const float*)d_in[8];
  const float* v_b  = (const float*)d_in[9];
  const float* o_w  = (const float*)d_in[10];
  const float* o_b  = (const float*)d_in[11];
  const int*  winp  = (const int*)d_in[12];
  float* out = (float*)d_out;

  const int S = 2048, H = 4608, KVD = 512;
  char* p = (char*)d_ws;
  auto take = [&](size_t bytes) { char* q = p; p += bytes; return q; };
  __bf16* Xb   = (__bf16*)take((size_t)S * H * 2);
  __bf16* Wq   = (__bf16*)take((size_t)H * H * 2);
  __bf16* Wk   = (__bf16*)take((size_t)KVD * H * 2);
  __bf16* Wv   = (__bf16*)take((size_t)KVD * H * 2);
  __bf16* Wo   = (__bf16*)take((size_t)H * H * 2);
  __bf16* Qpre = (__bf16*)take((size_t)S * H * 2);
  __bf16* Qr   = (__bf16*)take((size_t)S * H * 2);
  __bf16* Kpre = (__bf16*)take((size_t)S * KVD * 2);
  __bf16* Kr   = (__bf16*)take((size_t)S * KVD * 2);
  __bf16* Vt   = (__bf16*)take((size_t)KVD * S * 2);   // transposed V: [512][2048]
  __bf16* Ab   = Qpre;  // pre-RoPE Q is dead once attention runs

  // one-time fp32 -> bf16 casts
  auto cast = [&](const float* src, __bf16* dst, size_t n) {
    int n4 = (int)(n / 4);
    cast_kernel<<<(n4 + 255) / 256, 256, 0, stream>>>(src, dst, n4);
  };
  cast(hs,  Xb, (size_t)S * H);
  cast(q_w, Wq, (size_t)H * H);
  cast(k_w, Wk, (size_t)KVD * H);
  cast(v_w, Wv, (size_t)KVD * H);
  cast(o_w, Wo, (size_t)H * H);

  // QKV projections (bf16 WMMA, fp32 accumulate); V written transposed
  gemm_wmma<__bf16, false><<<(S / 64) * (H / 64) / 4, 128, 0, stream>>>(
      Xb, Wq, q_b, Qpre, S, H, H);
  gemm_wmma<__bf16, false><<<(S / 64) * (KVD / 64) / 4, 128, 0, stream>>>(
      Xb, Wk, k_b, Kpre, S, KVD, H);
  gemm_wmma<__bf16, true><<<(S / 64) * (KVD / 64) / 4, 128, 0, stream>>>(
      Xb, Wv, v_b, Vt, S, KVD, H);
  // RoPE
  rope_kernel<<<(S * H + 255) / 256, 256, 0, stream>>>(Qpre, Qr, cosT, sinT, 36, S * H);
  rope_kernel<<<(S * KVD + 255) / 256, 256, 0, stream>>>(Kpre, Kr, cosT, sinT, 4, S * KVD);
  // Sliding-window GQA flash attention
  dim3 ag(S / 16, 36);
  attn_kernel<<<ag, 32, 0, stream>>>(Qr, Kr, Vt, Ab, winp);
  // Output projection -> fp32 d_out
  gemm_wmma<float, false><<<(S / 64) * (H / 64) / 4, 128, 0, stream>>>(
      Ab, Wo, o_b, out, S, H, H);
}
